// LiquidTimeConstantNetwork_79250736545864
// MI455X (gfx1250) — matrix operand order repaired and checked by reading, not compiled, not run on hardware
//
#include <hip/hip_runtime.h>
#include <cstddef>

// LTC network: B=32, T=128, IN=128, H=256, OUT=128, STEPS=5, DT=0.01
//
// Pipeline (all on `stream`):
//   1) embed_gemm: emb[b*T+t][h] = X @ W_in^T + b_in          (WMMA f16)
//   2) ltc_scan:   persistent, 2 WGs x 512 thr (16 waves); both weight
//                  matrices held as per-wave B-fragment REGISTER arrays
//                  (loop-invariant, 128 VGPR/lane); double-buffered f16
//                  h-probe in LDS; 1 barrier + 16 wmma per wave per ODE eval
//   3) out_gemm:   out[b][t][o] = h_t @ W_out^T + b_out       (WMMA f16)
//
// Workspace layout: [0,4MB) emb f32 [B*T][H]; [4MB,8MB) hseq f32 [T][B][H].

typedef __attribute__((ext_vector_type(16))) _Float16 v16h;
typedef __attribute__((ext_vector_type(8)))  _Float16 v8h;
typedef __attribute__((ext_vector_type(8)))  float    v8f;

#define HDIM 256
#define BATCH 32
#define TSEQ 128
#define INDIM 128
#define OUTDIM 128
#define DT 0.01f
#define LDSS 264   // padded f16 row stride: 528B -> 4-bank row advance -> optimal 2-way LDS access

static __device__ __forceinline__ v16h cat8(v8h lo, v8h hi) {
  v16h r;
#pragma unroll
  for (int i = 0; i < 8; ++i) { r[i] = lo[i]; r[8 + i] = hi[i]; }
  return r;
}

static __device__ __forceinline__ v8f wmma_f16(v16h a, v16h b, v8f c) {
  return __builtin_amdgcn_wmma_f32_16x16x32_f16(false, a, false, b, (short)0, c, false, false);
}

// ---------------------------------------------------------------- embedding GEMM
// emb[m][n] = sum_k X[m][k] * Win[n][k] + bin[n]; m = b*T+t (4096), K=128, N=256
__global__ void embed_gemm(const float* __restrict__ X, const float* __restrict__ Win,
                           const float* __restrict__ bin, float* __restrict__ emb) {
  int wave = (blockIdx.x * blockDim.x + threadIdx.x) >> 5;   // 0..4095
  int lane = threadIdx.x & 31;
  int tm = wave >> 4, tn = wave & 15;                        // 256 x 16 tiles
  int nloc = lane & 15, lh = lane >> 4;
  v8f acc = {};
#pragma unroll
  for (int kt = 0; kt < 4; ++kt) {
    int k0 = kt * 32;
    v16h a, b;
    const float* ap = X + (size_t)(tm * 16 + nloc) * INDIM + k0 + lh * 16;
#pragma unroll
    for (int i = 0; i < 8; ++i) { a[i] = (_Float16)ap[i]; a[8 + i] = (_Float16)ap[8 + i]; }
    const float* bp = Win + (size_t)(tn * 16 + nloc) * INDIM + k0 + lh * 16;
#pragma unroll
    for (int i = 0; i < 16; ++i) b[i] = (_Float16)bp[i];
    acc = wmma_f16(a, b, acc);
  }
  int n = tn * 16 + nloc;
  float bias = bin[n];
#pragma unroll
  for (int r = 0; r < 8; ++r) {
    int m = tm * 16 + r + 8 * lh;
    emb[(size_t)m * HDIM + n] = acc[r] + bias;
  }
}

// ---------------------------------------------------------------- persistent LTC scan
// grid = 2 WGs (16 batch rows each), 512 threads = 16 waves; wave w owns N-tile w.
__global__ __launch_bounds__(512, 1) void ltc_scan(
    const float* __restrict__ emb, const float* __restrict__ Wrec,
    const float* __restrict__ Wadp, const float* __restrict__ tau,
    float* __restrict__ hseq) {
  __shared__ _Float16 lds_h[2][16 * LDSS];       // 16.9 KB double buffer, f16(h_probe)

  const int tid = threadIdx.x;
  const int wave = tid >> 5, lane = tid & 31;
  const int nloc = lane & 15, lh = lane >> 4;
  const int n = wave * 16 + nloc;                // this lane's H column (= weight row)
  const int wg = blockIdx.x;                     // batch half

  // Loop-invariant B-fragments for both weight matrices -> registers.
  // B layout (32x16 f16): lane holds 16 contiguous K at k0 + 16*(lane>=16), N = lane&15.
  v16h brec_f[8], badp_f[8];
#pragma unroll
  for (int kt = 0; kt < 8; ++kt) {
    int k0 = kt * 32 + lh * 16;
    const float* rp = Wrec + (size_t)n * HDIM + k0;
    const float* gp = Wadp + (size_t)n * HDIM + k0;
    v16h br, bg;
#pragma unroll
    for (int i = 0; i < 16; ++i) { br[i] = (_Float16)rp[i]; bg[i] = (_Float16)gp[i]; }
    brec_f[kt] = br;
    badp_f[kt] = bg;
  }

  float h_m[8], kcur[8], ksum[8], x_f[8];
#pragma unroll
  for (int r = 0; r < 8; ++r) { h_m[r] = 0.0f; kcur[r] = 0.0f; }
#pragma unroll
  for (int r = 0; r < 8; ++r) lds_h[0][(r + 8 * lh) * LDSS + n] = (_Float16)0.0f;
  const float tau_s = tau[n];
  const float hdt = 0.5f * DT;
  __syncthreads();

  int rb = 0;
  for (int t = 0; t < TSEQ; ++t) {
#pragma unroll
    for (int r = 0; r < 8; ++r) {
      int b = wg * 16 + r + 8 * lh;
      x_f[r] = emb[((size_t)b * TSEQ + t) * HDIM + n];
    }
    for (int it = 0; it < 5; ++it) {
#pragma unroll
      for (int r = 0; r < 8; ++r) ksum[r] = 0.0f;
      for (int e = 0; e < 4; ++e) {
        v8f crec = {}; v8f cadp = {};
#pragma unroll
        for (int kt = 0; kt < 8; ++kt) {
          int k0 = kt * 32;
          // A fragment: h_probe (f16) from LDS, shared by both GEMMs.
          // A layout (16x32 f16): lane holds 16 contiguous K at k0 + 16*(lane>=16), M = lane&15.
          const _Float16* ap = &lds_h[rb][nloc * LDSS + k0 + lh * 16];
          v16h a = cat8(*(const v8h*)ap, *(const v8h*)(ap + 8));
          crec = wmma_f16(a, brec_f[kt], crec);
          cadp = wmma_f16(a, badp_f[kt], cadp);
        }
        const float cprev = (e == 0) ? 0.0f : ((e == 3) ? DT : hdt);
        const float wgt   = (e == 1 || e == 2) ? 2.0f : 1.0f;
        float kv[8];
#pragma unroll
        for (int r = 0; r < 8; ++r) {
          float harg = h_m[r] + cprev * kcur[r];          // f32 ODE argument
          float S = x_f[r] + crec[r];
          float z = S + cadp[r];
          float sg = 1.0f / (1.0f + __expf(-z));
          float tvc = fminf(fmaxf(tau_s * sg, 1e-6f), 1e6f);
          kv[r] = (S - harg) / tvc;
          ksum[r] += wgt * kv[r];
          kcur[r] = kv[r];
        }
        if (e < 3) {
          const float cn = (e == 2) ? DT : hdt;
#pragma unroll
          for (int r = 0; r < 8; ++r)
            lds_h[rb ^ 1][(r + 8 * lh) * LDSS + n] = (_Float16)(h_m[r] + cn * kv[r]);
        } else {
#pragma unroll
          for (int r = 0; r < 8; ++r) {
            float d = (DT / 6.0f) * ksum[r];
            d = fminf(fmaxf(d, -1e6f), 1e6f);
            h_m[r] += d;
            lds_h[rb ^ 1][(r + 8 * lh) * LDSS + n] = (_Float16)h_m[r];
          }
        }
        __syncthreads();     // publish next h_probe, retire reads of current buffer
        rb ^= 1;
      }
    }
    // emit h_t for the output GEMM: hseq[t][b][h]
#pragma unroll
    for (int r = 0; r < 8; ++r) {
      int m = wg * 16 + r + 8 * lh;
      hseq[((size_t)t * BATCH + m) * HDIM + n] = h_m[r];
    }
  }
}

// ---------------------------------------------------------------- output GEMM
// out[b][t][o] = sum_h hseq[t*32+b][h] * Wout[o][h] + bout[o]; M=4096, K=256, N=128
__global__ void out_gemm(const float* __restrict__ Hs, const float* __restrict__ Wout,
                         const float* __restrict__ bout, float* __restrict__ out) {
  int wave = (blockIdx.x * blockDim.x + threadIdx.x) >> 5;   // 0..2047
  int lane = threadIdx.x & 31;
  int tm = wave >> 3, tn = wave & 7;                         // 256 x 8 tiles
  int nloc = lane & 15, lh = lane >> 4;
  v8f acc = {};
#pragma unroll
  for (int kt = 0; kt < 8; ++kt) {
    int k0 = kt * 32;
    v16h a, b;
    const float* ap = Hs + (size_t)(tm * 16 + nloc) * HDIM + k0 + lh * 16;
#pragma unroll
    for (int i = 0; i < 8; ++i) { a[i] = (_Float16)ap[i]; a[8 + i] = (_Float16)ap[8 + i]; }
    const float* bp = Wout + (size_t)(tn * 16 + nloc) * HDIM + k0 + lh * 16;
#pragma unroll
    for (int i = 0; i < 16; ++i) b[i] = (_Float16)bp[i];
    acc = wmma_f16(a, b, acc);
  }
  int o = tn * 16 + nloc;
  float bias = bout[o];
#pragma unroll
  for (int r = 0; r < 8; ++r) {
    int m = tm * 16 + r + 8 * lh;          // m = t*32 + b
    int tt = m >> 5, bb = m & 31;
    out[((size_t)bb * TSEQ + tt) * OUTDIM + o] = acc[r] + bias;
  }
}

// ---------------------------------------------------------------- launch
extern "C" void kernel_launch(void* const* d_in, const int* in_sizes, int n_in,
                              void* d_out, int out_size, void* d_ws, size_t ws_size,
                              hipStream_t stream) {
  (void)in_sizes; (void)n_in; (void)out_size; (void)ws_size;
  const float* X    = (const float*)d_in[0];  // [32][128][128]
  const float* Win  = (const float*)d_in[1];  // [256][128]
  const float* bin  = (const float*)d_in[2];  // [256]
  const float* Wrec = (const float*)d_in[3];  // [256][256]
  const float* Wadp = (const float*)d_in[4];  // [256][256]
  const float* Wout = (const float*)d_in[5];  // [128][256]
  const float* bout = (const float*)d_in[6];  // [128]
  const float* tau  = (const float*)d_in[7];  // [256]
  float* out = (float*)d_out;                 // [32][128][128]

  char* ws = (char*)d_ws;
  float* emb  = (float*)ws;                              // 4 MB
  float* hseq = (float*)(ws + 4u * 1024u * 1024u);       // 4 MB

  embed_gemm<<<1024, 128, 0, stream>>>(X, Win, bin, emb);     // 4096 waves
  ltc_scan<<<2, 512, 0, stream>>>(emb, Wrec, Wadp, tau, hseq);
  out_gemm<<<512, 128, 0, stream>>>(hseq, Wout, bout, out);   // 2048 waves
}
